// CrossAttention_57526791963210
// MI455X (gfx1250) — matrix operand order repaired and checked by reading, not compiled, not run on hardware
//
#include <hip/hip_runtime.h>

// ---------------------------------------------------------------------------
// CrossAttention for MI455X (gfx1250): bf16 WMMA 16x16x32 everywhere,
// TDM (tensor_load_to_lds) staging for the attention V blocks.
// B=4, T=26, L=256, C=128, H=4, dh=32.
// ---------------------------------------------------------------------------

typedef __bf16 bf16_t;
typedef __attribute__((ext_vector_type(16))) __bf16 v16bf;
typedef __attribute__((ext_vector_type(8)))  float  v8f;
typedef __attribute__((ext_vector_type(4)))  unsigned int u32x4;
typedef __attribute__((ext_vector_type(8)))  int          i32x8;
typedef __attribute__((ext_vector_type(4)))  int          i32x4;

#define NVIEW   104      // B*T
#define T_VIEWS 26
#define L_TOK   256
#define C_DIM   128
#define QKV_W   384
#define N_HEAD  4
#define DH      32

// half-wave (16 lane) row reductions matching the 16x16 C/D layout
__device__ __forceinline__ float halfrow_max(float v) {
    v = fmaxf(v, __shfl_xor(v, 1, 32));
    v = fmaxf(v, __shfl_xor(v, 2, 32));
    v = fmaxf(v, __shfl_xor(v, 4, 32));
    v = fmaxf(v, __shfl_xor(v, 8, 32));
    return v;
}
__device__ __forceinline__ float halfrow_sum(float v) {
    v += __shfl_xor(v, 1, 32);
    v += __shfl_xor(v, 2, 32);
    v += __shfl_xor(v, 4, 32);
    v += __shfl_xor(v, 8, 32);
    return v;
}

// ---------------------------------------------------------------------------
// TDM: DMA a 2D tile of 2-byte elements (tile_w x tile_h, row stride in
// elements) from global memory into LDS at lds_byte_off.
// D# layout per CDNA5 ISA ch.8: group0 = {count, lds_addr, global_addr, type},
// group1 = {data_size, tensor_dim0/1, tile_dim0/1/2, dim0_stride}.
// ---------------------------------------------------------------------------
#if __has_builtin(__builtin_amdgcn_tensor_load_to_lds)
#define HAVE_TDM 1
__device__ __forceinline__ void tdm_load_2d_to_lds(unsigned lds_byte_off,
                                                   const void* gaddr,
                                                   unsigned tile_w_elems,
                                                   unsigned tile_h_rows,
                                                   unsigned row_stride_elems) {
    unsigned long long ga = (unsigned long long)(__SIZE_TYPE__)gaddr;
    u32x4 g0;
    g0[0] = 1u;                                                    // count=1, user mode
    g0[1] = lds_byte_off;                                          // lds_addr (bytes)
    g0[2] = (unsigned)(ga & 0xFFFFFFFFu);                          // global_addr[31:0]
    g0[3] = (unsigned)((ga >> 32) & 0x01FFFFFFu) | (0x2u << 30);   // addr[56:32] | type=2
    i32x8 g1;
    g1[0] = (int)(1u << 16);                                       // data_size=1 -> 2 bytes
    g1[1] = (int)((tile_w_elems & 0xFFFFu) << 16);                 // tensor_dim0[15:0]
    g1[2] = (int)((tile_h_rows & 0xFFFFu) << 16);                  // tensor_dim1[15:0]
    g1[3] = (int)((tile_w_elems & 0xFFFFu) << 16);                 // tile_dim0
    g1[4] = (int)(tile_h_rows & 0xFFFFu);                          // tile_dim1 (tile_dim2=0)
    g1[5] = (int)row_stride_elems;                                 // tensor_dim0_stride[31:0]
    g1[6] = 0;
    g1[7] = 0;
    i32x4 gz; gz[0] = 0; gz[1] = 0; gz[2] = 0; gz[3] = 0;
#if defined(__clang_major__) && (__clang_major__ >= 23)
    i32x8 gz8;
#pragma unroll
    for (int i = 0; i < 8; ++i) gz8[i] = 0;
    __builtin_amdgcn_tensor_load_to_lds(g0, g1, gz, gz, gz8, 0);
#else
    __builtin_amdgcn_tensor_load_to_lds(g0, g1, gz, gz, 0);
#endif
}
#else
#define HAVE_TDM 0
#endif

// ---------------------------------------------------------------------------
// Kernel 0: build the neighbor table (replicates _neighbors()) on device.
// ---------------------------------------------------------------------------
__global__ void neigh_kernel(int* __restrict__ idxTab, int* __restrict__ mskTab) {
    if (threadIdx.x != 0 || blockIdx.x != 0) return;
    int ax[26], ay[26];
    ax[0] = 0; ay[0] = 90; ax[1] = 0; ay[1] = -90;
    int n = 2;
    for (int i = 1; i < 4; ++i)
        for (int j = 0; j < 8; ++j) { ax[n] = 45 * j; ay[n] = -90 + 45 * i; ++n; }
    bool ab[26][26];
    for (int i = 0; i < 26; ++i)
        for (int j = 0; j < 26; ++j) {
            int dx = ax[i] - ax[j]; if (dx < 0) dx = -dx; if (360 - dx < dx) dx = 360 - dx;
            int dy = ay[i] - ay[j]; if (dy < 0) dy = -dy; if (360 - dy < dy) dy = 360 - dy;
            ab[i][j] = ((dx + dy) == 45);
        }
    for (int k = 18; k < 26; k += 2) { ab[0][k] = true; ab[k][0] = true; }
    for (int k = 2;  k < 10; k += 2) { ab[1][k] = true; ab[k][1] = true; }
    for (int i = 0; i < 26; ++i) {
        int cnt = 0;
        for (int j = 0; j < 26 && cnt < 4; ++j)
            if (ab[i][j]) { idxTab[i * 4 + cnt] = j; mskTab[i * 4 + cnt] = 1; ++cnt; }
        for (; cnt < 4; ++cnt) { idxTab[i * 4 + cnt] = 0; mskTab[i * 4 + cnt] = 0; }
    }
}

// ---------------------------------------------------------------------------
// Kernel 1: x [view][C][L] f32  ->  xt [view][L][C] bf16
// ---------------------------------------------------------------------------
__global__ void xcvt_kernel(const float* __restrict__ x, bf16_t* __restrict__ xt) {
    int i = blockIdx.x * 256 + threadIdx.x;
    if (i >= NVIEW * C_DIM * L_TOK) return;
    int view = i >> 15;          // / (128*256)
    int rem  = i & 32767;
    int c    = rem >> 8;         // / 256
    int l    = rem & 255;
    xt[(size_t)view * (L_TOK * C_DIM) + l * C_DIM + c] = (bf16_t)x[i];
}

// ---------------------------------------------------------------------------
// Kernel 2: weights -> transposed bf16 copies (WMMA B-operand friendly)
// ---------------------------------------------------------------------------
__global__ void wcvt_kernel(const float* __restrict__ w_in, const float* __restrict__ w_out,
                            bf16_t* __restrict__ wT, bf16_t* __restrict__ wOutT) {
    int i = blockIdx.x * 256 + threadIdx.x;
    if (i < QKV_W * C_DIM) {
        int co = i >> 7, ci = i & 127;
        wT[co * C_DIM + ci] = (bf16_t)w_in[i];
    }
    int j = i - QKV_W * C_DIM;
    if (j >= 0 && j < C_DIM * C_DIM) {
        int co = j >> 7, ci = j & 127;
        wOutT[co * C_DIM + ci] = (bf16_t)w_out[j];
    }
}

// ---------------------------------------------------------------------------
// Kernel 3: QKV projection.  qkv[view][L][384] = xt[view] @ w_inT + b_in
// ---------------------------------------------------------------------------
__global__ __launch_bounds__(256) void qkv_kernel(const bf16_t* __restrict__ xt,
                                                  const bf16_t* __restrict__ wT,
                                                  const float*  __restrict__ b_in,
                                                  bf16_t* __restrict__ qkv) {
    const int view  = blockIdx.x;
    const int tid   = threadIdx.x;
    const int wv    = tid >> 5;
    const int lane  = tid & 31;
    const int lrow  = lane & 15;
    const int lhalf = lane >> 4;
    const int kb    = lhalf * 8;

    const bf16_t* xbase = xt + (size_t)view * (L_TOK * C_DIM);
    bf16_t*       obase = qkv + (size_t)view * (L_TOK * QKV_W);

    for (int tile = wv; tile < 16 * 24; tile += 8) {
        const int mt = tile / 24;
        const int nt = tile - mt * 24;
        const float bias = b_in[nt * 16 + lrow];
        v8f acc;
#pragma unroll
        for (int r = 0; r < 8; ++r) acc[r] = bias;
#pragma unroll
        for (int kc = 0; kc < 4; ++kc) {
            v16bf a, bm;
            const bf16_t* ap = xbase + (mt * 16 + lrow) * C_DIM + kc * 32;
            ((uint4*)&a)[0] = *(const uint4*)(ap + kb);
            ((uint4*)&a)[1] = *(const uint4*)(ap + kb + 16);
            const bf16_t* bp = wT + (size_t)(nt * 16 + lrow) * C_DIM + kc * 32;
            ((uint4*)&bm)[0] = *(const uint4*)(bp + kb);
            ((uint4*)&bm)[1] = *(const uint4*)(bp + kb + 16);
            acc = __builtin_amdgcn_wmma_f32_16x16x32_bf16(false, a, false, bm,
                                                          (short)0, acc, false, false);
        }
#pragma unroll
        for (int r = 0; r < 8; ++r) {
            int row = mt * 16 + r + 8 * lhalf;
            obase[row * QKV_W + nt * 16 + lrow] = (bf16_t)acc[r];
        }
    }
}

// ---------------------------------------------------------------------------
// Kernel 4: flash attention per (view, head). 8 waves; each wave owns two
// 16-row Q tiles. Per neighbor view:
//   vs[key][dh] staged by the Tensor Data Mover (async, TENSORcnt),
//   kt[dh][key] staged by vectorized b128 loads + b16 LDS scatter (transpose).
// Masked (invalid) neighbors are skipped (== the -1e9 bias exactly).
// ---------------------------------------------------------------------------
__global__ __launch_bounds__(256) void attn_kernel(const bf16_t* __restrict__ qkv,
                                                   bf16_t* __restrict__ attn,
                                                   const int* __restrict__ idxTab,
                                                   const int* __restrict__ mskTab) {
    __shared__ __align__(16) bf16_t kt[DH][L_TOK];       // 16 KB  K^T
    __shared__ __align__(16) bf16_t vs[L_TOK][DH];       // 16 KB  V
    __shared__ __align__(16) bf16_t pls[8][16][32];      //  8 KB  per-wave P scratch

    const int view  = blockIdx.x;            // 0..103
    const int h     = blockIdx.y;            // 0..3
    const int b     = view / T_VIEWS;
    const int t     = view - b * T_VIEWS;
    const int tid   = threadIdx.x;
    const int wv    = tid >> 5;
    const int lane  = tid & 31;
    const int lrow  = lane & 15;
    const int lhalf = lane >> 4;
    const int kb    = lhalf * 8;
    const float sc  = 0.17677669529663687f; // 1/sqrt(32)

    v16bf aq[2];
    v8f acc0[2], acc1[2];
    float m[2][8], lsum[2][8];
#pragma unroll
    for (int qi = 0; qi < 2; ++qi) {
        int qrow = (wv * 2 + qi) * 16 + lrow;
        const bf16_t* qp = qkv + (size_t)view * (L_TOK * QKV_W) + (size_t)qrow * QKV_W + h * DH;
        ((uint4*)&aq[qi])[0] = *(const uint4*)(qp + kb);
        ((uint4*)&aq[qi])[1] = *(const uint4*)(qp + kb + 16);
#pragma unroll
        for (int r = 0; r < 8; ++r) {
            m[qi][r] = -1e30f; lsum[qi][r] = 0.0f;
            acc0[qi][r] = 0.0f; acc1[qi][r] = 0.0f;
        }
    }

    for (int kn = 0; kn < 4; ++kn) {
        if (!mskTab[t * 4 + kn]) continue;          // uniform across the block
        const int nview = b * T_VIEWS + idxTab[t * 4 + kn];
        const bf16_t* nb = qkv + (size_t)nview * (L_TOK * QKV_W);
        if (kn < 3 && mskTab[t * 4 + kn + 1]) {     // prefetch next neighbor's block
            const bf16_t* pf = qkv + (size_t)(b * T_VIEWS + idxTab[t * 4 + kn + 1]) * (L_TOK * QKV_W);
            __builtin_prefetch(pf + C_DIM + h * DH, 0, 0);
        }
        __syncthreads();                            // protect LDS reuse

#if HAVE_TDM
        // V block: 256 rows x 32 (2B) elements, global row stride 384 elems.
        if (wv == 0) {
            tdm_load_2d_to_lds((unsigned)(__SIZE_TYPE__)&vs[0][0],
                               nb + 2 * C_DIM + h * DH, DH, L_TOK, QKV_W);
        }
#else
        for (int e = tid; e < L_TOK * DH; e += 256) {
            int key = e >> 5, dh = e & 31;
            vs[key][dh] = nb[key * QKV_W + 2 * C_DIM + h * DH + dh];
        }
#endif
        // K^T block: vectorized b128 row loads, b16 LDS transpose scatter.
        for (int e = tid; e < L_TOK * 4; e += 256) {
            int key = e >> 2, dhb = (e & 3) * 8;
            uint4 d = *(const uint4*)(nb + key * QKV_W + C_DIM + h * DH + dhb);
            const bf16_t* dv = (const bf16_t*)&d;
#pragma unroll
            for (int i = 0; i < 8; ++i) kt[dhb + i][key] = dv[i];
        }
#if HAVE_TDM
        if (wv == 0) __builtin_amdgcn_s_wait_tensorcnt(0);
#endif
        __syncthreads();

#pragma unroll
        for (int qi = 0; qi < 2; ++qi) {
            for (int ck = 0; ck < 8; ++ck) {        // 32 keys per chunk
                const int keyb = ck * 32;
                // Score B operands: two 32(dh) x 16(key) tiles; lane%16 = key (N index).
                v16bf bk0, bk1;
#pragma unroll
                for (int s = 0; s < 16; ++s) {
                    int kk = kb + (s & 7) + ((s >> 3) << 4);
                    bk0[s] = kt[kk][keyb + lrow];
                    bk1[s] = kt[kk][keyb + 16 + lrow];
                }
                v8f zero;
#pragma unroll
                for (int r = 0; r < 8; ++r) zero[r] = 0.0f;
                v8f s0 = __builtin_amdgcn_wmma_f32_16x16x32_bf16(false, aq[qi], false, bk0,
                                                                 (short)0, zero, false, false);
                v8f s1 = __builtin_amdgcn_wmma_f32_16x16x32_bf16(false, aq[qi], false, bk1,
                                                                 (short)0, zero, false, false);
                // online softmax (rows live on 16-lane halves of the C/D layout)
                float p0a[8], p1a[8];
#pragma unroll
                for (int r = 0; r < 8; ++r) {
                    float x0 = s0[r] * sc, x1 = s1[r] * sc;
                    float cm = halfrow_max(fmaxf(x0, x1));
                    float nm = fmaxf(m[qi][r], cm);
                    float corr = __expf(m[qi][r] - nm);
                    m[qi][r] = nm;
                    float e0 = __expf(x0 - nm);
                    float e1 = __expf(x1 - nm);
                    lsum[qi][r] = lsum[qi][r] * corr + halfrow_sum(e0 + e1);
                    acc0[qi][r] *= corr;
                    acc1[qi][r] *= corr;
                    p0a[r] = e0; p1a[r] = e1;
                }
                // C/D layout -> A layout via per-wave LDS bounce
#pragma unroll
                for (int r = 0; r < 8; ++r) {
                    int prow = r + 8 * lhalf;
                    pls[wv][prow][lrow]      = (bf16_t)p0a[r];
                    pls[wv][prow][16 + lrow] = (bf16_t)p1a[r];
                }
                asm volatile("s_wait_dscnt 0" ::: "memory");
                __builtin_amdgcn_wave_barrier();
                v16bf ap;
                ((uint4*)&ap)[0] = *(const uint4*)&pls[wv][lrow][kb];
                ((uint4*)&ap)[1] = *(const uint4*)&pls[wv][lrow][kb + 16];
                // PV B operands: 32(key) x 16(dh) tiles; lane%16 = dh (N index).
                v16bf bv0, bv1;
#pragma unroll
                for (int s = 0; s < 16; ++s) {
                    int kk = kb + (s & 7) + ((s >> 3) << 4);
                    bv0[s] = vs[keyb + kk][lrow];
                    bv1[s] = vs[keyb + kk][16 + lrow];
                }
                acc0[qi] = __builtin_amdgcn_wmma_f32_16x16x32_bf16(false, ap, false, bv0,
                                                                   (short)0, acc0[qi], false, false);
                acc1[qi] = __builtin_amdgcn_wmma_f32_16x16x32_bf16(false, ap, false, bv1,
                                                                   (short)0, acc1[qi], false, false);
            }
        }
    }

    // finalize: out = acc / lsum, write bf16 [view][L][C]
#pragma unroll
    for (int qi = 0; qi < 2; ++qi) {
#pragma unroll
        for (int r = 0; r < 8; ++r) {
            int row = (wv * 2 + qi) * 16 + r + 8 * lhalf;
            float inv = 1.0f / lsum[qi][r];
            bf16_t* op = attn + (size_t)view * (L_TOK * C_DIM) + row * C_DIM + h * DH;
            op[lrow]      = (bf16_t)(acc0[qi][r] * inv);
            op[16 + lrow] = (bf16_t)(acc1[qi][r] * inv);
        }
    }
}

// ---------------------------------------------------------------------------
// Kernel 5: output projection + transpose back to [view][C][L] f32.
// ---------------------------------------------------------------------------
__global__ __launch_bounds__(256) void out_kernel(const bf16_t* __restrict__ attn,
                                                  const bf16_t* __restrict__ wOutT,
                                                  const float*  __restrict__ b_out,
                                                  float* __restrict__ out) {
    const int view  = blockIdx.x;
    const int tid   = threadIdx.x;
    const int wv    = tid >> 5;
    const int lane  = tid & 31;
    const int lrow  = lane & 15;
    const int lhalf = lane >> 4;
    const int kb    = lhalf * 8;

    const bf16_t* abase = attn + (size_t)view * (L_TOK * C_DIM);
    float*        obase = out + (size_t)view * (C_DIM * L_TOK);

    for (int tile = wv; tile < 16 * 8; tile += 8) {
        const int mt = tile >> 3;
        const int nt = tile & 7;
        const float bias = b_out[nt * 16 + lrow];
        v8f acc;
#pragma unroll
        for (int r = 0; r < 8; ++r) acc[r] = bias;
#pragma unroll
        for (int kc = 0; kc < 4; ++kc) {
            v16bf a, bm;
            const bf16_t* ap = abase + (mt * 16 + lrow) * C_DIM + kc * 32;
            ((uint4*)&a)[0] = *(const uint4*)(ap + kb);
            ((uint4*)&a)[1] = *(const uint4*)(ap + kb + 16);
            const bf16_t* bp = wOutT + (size_t)(nt * 16 + lrow) * C_DIM + kc * 32;
            ((uint4*)&bm)[0] = *(const uint4*)(bp + kb);
            ((uint4*)&bm)[1] = *(const uint4*)(bp + kb + 16);
            acc = __builtin_amdgcn_wmma_f32_16x16x32_bf16(false, a, false, bm,
                                                          (short)0, acc, false, false);
        }
#pragma unroll
        for (int r = 0; r < 8; ++r) {
            int row = mt * 16 + r + 8 * lhalf;   // token index
            int col = nt * 16 + lrow;            // channel index
            obase[col * L_TOK + row] = acc[r];   // fused transpose to [C][L]
        }
    }
}

// ---------------------------------------------------------------------------
// Host-side launcher (graph-capture safe: async launches only, scratch in ws)
// ---------------------------------------------------------------------------
extern "C" void kernel_launch(void* const* d_in, const int* in_sizes, int n_in,
                              void* d_out, int out_size, void* d_ws, size_t ws_size,
                              hipStream_t stream) {
    (void)in_sizes; (void)n_in; (void)out_size; (void)ws_size;
    const float* x     = (const float*)d_in[0];
    const float* w_in  = (const float*)d_in[1];
    const float* b_in  = (const float*)d_in[2];
    const float* w_out = (const float*)d_in[3];
    const float* b_out = (const float*)d_in[4];
    float* out = (float*)d_out;

    char* ws = (char*)d_ws;
    // workspace layout (256B-aligned sections)
    int*    idxTab = (int*)(ws + 0);                         //   416 B
    int*    mskTab = (int*)(ws + 512);                       //   416 B
    bf16_t* wT     = (bf16_t*)(ws + 1024);                   // 98304 B
    bf16_t* wOutT  = (bf16_t*)(ws + 99328);                  // 32768 B
    bf16_t* xt     = (bf16_t*)(ws + 132096);                 // 6.50 MB
    bf16_t* qkv    = (bf16_t*)(ws + 6947840);                // 19.5 MB
    bf16_t* attn   = (bf16_t*)(ws + 27395072);               // 6.50 MB

    neigh_kernel<<<1, 1, 0, stream>>>(idxTab, mskTab);
    wcvt_kernel<<<(QKV_W * C_DIM + C_DIM * C_DIM + 255) / 256, 256, 0, stream>>>(
        w_in, w_out, wT, wOutT);
    xcvt_kernel<<<(NVIEW * C_DIM * L_TOK + 255) / 256, 256, 0, stream>>>(x, xt);
    qkv_kernel<<<NVIEW, 256, 0, stream>>>(xt, wT, b_in, qkv);
    attn_kernel<<<dim3(NVIEW, N_HEAD), 256, 0, stream>>>(qkv, attn, idxTab, mskTab);
    out_kernel<<<NVIEW, 256, 0, stream>>>(attn, wOutT, b_out, out);
}
